// HeteroGNN_11003706212418
// MI455X (gfx1250) — matrix operand order repaired and checked by reading, not compiled, not run on hardware
//
#include <hip/hip_runtime.h>
#include <hip/hip_bf16.h>
#include <stdint.h>

// ---------------------------------------------------------------------------
// CDNA5 (gfx1250) WMMA / TDM types
// ---------------------------------------------------------------------------
typedef __attribute__((ext_vector_type(16))) __bf16 v16bf;
typedef __attribute__((ext_vector_type(8)))  float  v8f;
typedef __attribute__((ext_vector_type(4)))  unsigned int u32x4;
typedef __attribute__((ext_vector_type(8)))  unsigned int u32x8;

union FragU {
  uint4 q[2];   // 32 bytes
  v16bf v;      // 16 x bf16
};

__device__ __forceinline__ unsigned short f2bf(float f) {
  unsigned int u = __builtin_bit_cast(unsigned int, f);
  u += 0x7FFFu + ((u >> 16) & 1u);   // round-to-nearest-even
  return (unsigned short)(u >> 16);
}

#define LDS_PITCH 136   // 128 + 8 pad: 16B-aligned fragments, spreads LDS banks

// ---------------------------------------------------------------------------
// Tensor Data Mover: DMA a 128-col bf16 2D tile (tile_dim1=128 rows) from
// global into LDS with hardware row padding (128 elem row + 16B pad == pitch
// 136 ushorts). rows_avail = tensor extent in dim1; OOB rows read as zero
// (free tail handling). Descriptor packing per CDNA5 ISA 8.3-8.5:
//   g0: count=1 | lds_addr | global_addr[56:0] | type=2
//   g1: data_size=2B, pad_enable, pad_interval=5 (64 DW = 1 row),
//       pad_amount=3 (4 DW = 16B), tensor_dim0=128, tensor_dim1=rows_avail,
//       tile_dim0=128, tile_dim1=128, dim0_stride=128
//   g2/g3: zero (tile_dim2/3 = 0 -> unused; 4-group form always accepted)
// Issued once per workgroup (wave 0); TDM ignores EXEC. Tracked by TENSORcnt.
// ---------------------------------------------------------------------------
__device__ __forceinline__ void tdm_load_tile_bf16(const unsigned short* gsrc,
                                                   unsigned lds_off,
                                                   unsigned rows_avail) {
  unsigned long long ga = (unsigned long long)(uintptr_t)gsrc;
  u32x4 g0;
  g0[0] = 1u;                                              // count=1 (user D#)
  g0[1] = lds_off;                                         // lds_addr (bytes)
  g0[2] = (unsigned)ga;                                    // global_addr[31:0]
  g0[3] = (unsigned)((ga >> 32) & 0x01FFFFFFu) | (2u << 30); // addr[56:32]|type=2
  u32x8 g1;
  g1[0] = (1u << 16) | (1u << 20) | (5u << 22) | (3u << 25);
  g1[1] = 128u << 16;                                      // tensor_dim0 lo16
  g1[2] = (rows_avail & 0xFFFFu) << 16;                    // dim0 hi | dim1 lo
  g1[3] = (rows_avail >> 16) | (128u << 16);               // dim1 hi | tile_dim0
  g1[4] = 128u;                                            // tile_dim1 | tile_dim2=0
  g1[5] = 128u;                                            // dim0_stride lo32
  g1[6] = 0u;                                              // stride hi | dim1_stride lo
  g1[7] = 0u;
  u32x4 gz = (u32x4)0u;
  asm volatile("tensor_load_to_lds %0, %1, %2, %3"
               :: "s"(g0), "s"(g1), "s"(gz), "s"(gz)
               : "memory");
}

// ---------------------------------------------------------------------------
// Utility kernels
// ---------------------------------------------------------------------------
__global__ void k_zero(float* __restrict__ p, int n) {
  int gid = blockIdx.x * 256 + threadIdx.x;
  if (gid < n) p[gid] = 0.f;
}

// Convert 128x128 f32 row-major [k][n] mats into bf16 transposed [n][k] blocks.
__global__ void k_convT(const float* __restrict__ in, unsigned short* __restrict__ out) {
  int mb  = blockIdx.x >> 6;
  int idx = (blockIdx.x & 63) * 256 + threadIdx.x;  // 0..16383 = k*128+n
  int k = idx >> 7, n = idx & 127;
  out[(size_t)mb * 16384 + n * 128 + k] = f2bf(in[(size_t)mb * 16384 + idx]);
}

// One wave per edge: gather src row (4 f32/lane), scatter-add into agg[dst].
__global__ void k_scatter(const float* __restrict__ X, const int* __restrict__ ei,
                          float* __restrict__ agg, float* __restrict__ cnt, int E) {
  int gid  = blockIdx.x * 256 + threadIdx.x;
  int e    = gid >> 5;
  int lane = gid & 31;
  if (e >= E) return;
  int src = ei[e];
  int dst = ei[E + e];
  const float4 v = *(const float4*)(X + (size_t)src * 128 + lane * 4);
  float* a = agg + (size_t)dst * 128 + lane * 4;
  atomicAdd(a + 0, v.x);
  atomicAdd(a + 1, v.y);
  atomicAdd(a + 2, v.z);
  atomicAdd(a + 3, v.w);
  if (lane == 0) atomicAdd(cnt + dst, 1.0f);
}

__global__ void k_finalize(float* __restrict__ agg, const float* __restrict__ cnt, int n128) {
  int gid = blockIdx.x * 256 + threadIdx.x;
  if (gid >= n128) return;
  float c = cnt[gid >> 7];
  agg[gid] *= (1.0f / fmaxf(c, 1.0f));
}

// ---------------------------------------------------------------------------
// GEMM 1: Out_bf16[N][128] = A_f32[N][128] @ W[128][128] + bias
// B tile DMA'd by TDM; A tile converted f32->bf16 by the waves in parallel.
// ---------------------------------------------------------------------------
__global__ __launch_bounds__(256)
void k_gemm_xw(const float* __restrict__ A, const unsigned short* __restrict__ Wt,
               const float* __restrict__ bias, unsigned short* __restrict__ Out, int nrows) {
  __shared__ unsigned short As[128 * LDS_PITCH];
  __shared__ unsigned short Bs[128 * LDS_PITCH];
  const int tid = threadIdx.x;
  const int rowBase = blockIdx.x * 128;

  if (tid < 32)   // wave 0 kicks the weight-tile DMA (overlaps A conversion)
    tdm_load_tile_bf16(Wt, (unsigned)(uintptr_t)(void*)Bs, 128u);

  for (int i = 0; i < 64; ++i) {
    int idx = tid + i * 256;
    int r = idx >> 7, c = idx & 127;
    int gr = rowBase + r;
    float v = (gr < nrows) ? A[(size_t)gr * 128 + c] : 0.f;
    As[r * LDS_PITCH + c] = f2bf(v);
  }
  if (tid < 32) __builtin_amdgcn_s_wait_tensorcnt(0);
  __syncthreads();

  const int lane = tid & 31;
  const int w    = tid >> 5;
  const int ml   = lane & 15;
  const int kbA  = (lane < 16) ? 0 : 8;    // A frag: K in {kb..kb+7, kb+16..kb+23}
  const int kbB  = (lane < 16) ? 0 : 8;    // B frag: K in {kb..kb+7, kb+16..kb+23}
  const unsigned short* arow = As + (w * 16 + ml) * LDS_PITCH;

  v8f acc[8];
  #pragma unroll
  for (int j = 0; j < 8; ++j)
    #pragma unroll
    for (int e = 0; e < 8; ++e) acc[j][e] = 0.f;

  #pragma unroll
  for (int ks = 0; ks < 128; ks += 32) {
    FragU fa;
    fa.q[0] = *(const uint4*)(arow + ks + kbA);
    fa.q[1] = *(const uint4*)(arow + ks + kbA + 16);
    #pragma unroll
    for (int j = 0; j < 8; ++j) {
      FragU fb;
      const unsigned short* bp = Bs + (j * 16 + ml) * LDS_PITCH + ks + kbB;
      fb.q[0] = *(const uint4*)(bp);
      fb.q[1] = *(const uint4*)(bp + 16);
      acc[j] = __builtin_amdgcn_wmma_f32_16x16x32_bf16(false, fa.v, false, fb.v,
                                                       (short)0, acc[j], false, false);
    }
  }

  const int gr0 = rowBase + w * 16 + ((lane < 16) ? 0 : 8);
  #pragma unroll
  for (int j = 0; j < 8; ++j) {
    int col = j * 16 + ml;
    float bv = bias[col];
    #pragma unroll
    for (int r = 0; r < 8; ++r) {
      int gr = gr0 + r;
      if (gr < nrows) Out[(size_t)gr * 128 + col] = f2bf(acc[j][r] + bv);
    }
  }
}

// ---------------------------------------------------------------------------
// GEMM 2: Out_f32[N][128] (+)= A1_bf16 @ W1 + A2_bf16 @ W2 + bias
// Both A and B tiles are bf16 in global, so both are DMA'd by TDM; the
// descriptor's tensor_dim1 clamps the tail block (OOB rows read zero).
// ---------------------------------------------------------------------------
__global__ __launch_bounds__(256)
void k_gemm2(const unsigned short* __restrict__ A1, const unsigned short* __restrict__ A2,
             const unsigned short* __restrict__ W1t, const unsigned short* __restrict__ W2t,
             const float* __restrict__ bias, float* __restrict__ Out, int nrows, int accum) {
  __shared__ unsigned short As[128 * LDS_PITCH];
  __shared__ unsigned short Bs[128 * LDS_PITCH];
  const int tid = threadIdx.x;
  const int rowBase = blockIdx.x * 128;
  const int lane = tid & 31;
  const int w    = tid >> 5;
  const int ml   = lane & 15;
  const int kbA  = (lane < 16) ? 0 : 8;
  const int kbB  = (lane < 16) ? 0 : 8;
  const unsigned rowsAvail = (unsigned)(nrows - rowBase);
  const unsigned asOff = (unsigned)(uintptr_t)(void*)As;
  const unsigned bsOff = (unsigned)(uintptr_t)(void*)Bs;

  v8f acc[8];
  #pragma unroll
  for (int j = 0; j < 8; ++j)
    #pragma unroll
    for (int e = 0; e < 8; ++e) acc[j][e] = 0.f;

  const unsigned short* Asrc[2] = {A1, A2};
  const unsigned short* Wsrc[2] = {W1t, W2t};

  for (int half = 0; half < 2; ++half) {
    __syncthreads();                 // previous compute done before restaging
    if (tid < 32) {
      tdm_load_tile_bf16(Asrc[half] + (size_t)rowBase * 128, asOff, rowsAvail);
      tdm_load_tile_bf16(Wsrc[half], bsOff, 128u);
      __builtin_amdgcn_s_wait_tensorcnt(0);
    }
    __syncthreads();

    const unsigned short* arow = As + (w * 16 + ml) * LDS_PITCH;
    #pragma unroll
    for (int ks = 0; ks < 128; ks += 32) {
      FragU fa;
      fa.q[0] = *(const uint4*)(arow + ks + kbA);
      fa.q[1] = *(const uint4*)(arow + ks + kbA + 16);
      #pragma unroll
      for (int j = 0; j < 8; ++j) {
        FragU fb;
        const unsigned short* bp = Bs + (j * 16 + ml) * LDS_PITCH + ks + kbB;
        fb.q[0] = *(const uint4*)(bp);
        fb.q[1] = *(const uint4*)(bp + 16);
        acc[j] = __builtin_amdgcn_wmma_f32_16x16x32_bf16(false, fa.v, false, fb.v,
                                                         (short)0, acc[j], false, false);
      }
    }
  }

  const int gr0 = rowBase + w * 16 + ((lane < 16) ? 0 : 8);
  #pragma unroll
  for (int j = 0; j < 8; ++j) {
    int col = j * 16 + ml;
    float bv = bias[col];
    #pragma unroll
    for (int r = 0; r < 8; ++r) {
      int gr = gr0 + r;
      if (gr < nrows) {
        size_t o = (size_t)gr * 128 + col;
        float v = acc[j][r] + bv;
        if (accum) v += Out[o];
        Out[o] = v;
      }
    }
  }
}

// ---------------------------------------------------------------------------
// BatchNorm (training stats) + LeakyReLU, with a pre-scale folded in
// ---------------------------------------------------------------------------
__global__ void k_bn_stats(const float* __restrict__ X, float* __restrict__ stats,
                           float scale, int nrows) {
  __shared__ float sh[512];
  int tid  = threadIdx.x;
  int col  = tid & 127;
  int half = tid >> 7;
  float s = 0.f, sq = 0.f;
  for (int r = blockIdx.x * 2 + half; r < nrows; r += gridDim.x * 2) {
    float v = scale * X[(size_t)r * 128 + col];
    s += v; sq += v * v;
  }
  sh[tid] = s; sh[256 + tid] = sq;
  __syncthreads();
  if (tid < 128) {
    atomicAdd(&stats[tid],       sh[tid] + sh[tid + 128]);
    atomicAdd(&stats[128 + tid], sh[256 + tid] + sh[256 + tid + 128]);
  }
}

__global__ void k_bn_apply(float* __restrict__ X, const float* __restrict__ stats,
                           const float* __restrict__ g, const float* __restrict__ b,
                           float scale, int nrows) {
  int gid = blockIdx.x * 256 + threadIdx.x;
  if (gid >= nrows * 128) return;
  int col = gid & 127;
  float inv_n = 1.0f / (float)nrows;
  float m   = stats[col] * inv_n;
  float var = stats[128 + col] * inv_n - m * m;
  float rs  = rsqrtf(var + 1e-5f);
  float v   = scale * X[gid];
  float y   = (v - m) * rs * g[col] + b[col];
  X[gid] = (y >= 0.f) ? y : 0.01f * y;
}

// Final [N,128] @ [128,8] + bias
__global__ void k_proj(const float* __restrict__ H, const float* __restrict__ Wp,
                       const float* __restrict__ bp, float* __restrict__ out, int nrows) {
  int gid = blockIdx.x * 256 + threadIdx.x;
  int row = gid >> 3, l = gid & 7;
  if (row >= nrows) return;
  float acc = bp[l];
  #pragma unroll 4
  for (int k = 0; k < 128; ++k) acc += H[(size_t)row * 128 + k] * Wp[k * 8 + l];
  out[(size_t)row * 8 + l] = acc;
}

// ---------------------------------------------------------------------------
// Host orchestration
// ---------------------------------------------------------------------------
extern "C" void kernel_launch(void* const* d_in, const int* in_sizes, int n_in,
                              void* d_out, int out_size, void* d_ws, size_t ws_size,
                              hipStream_t stream) {
  (void)n_in; (void)out_size; (void)ws_size;
  const float* x0     = (const float*)d_in[0];
  const float* x1     = (const float*)d_in[1];
  const float* Wn1    = (const float*)d_in[2];
  const float* Ws1    = (const float*)d_in[3];
  const float* Wu1    = (const float*)d_in[4];
  const float* Wn2    = (const float*)d_in[5];
  const float* Ws2    = (const float*)d_in[6];
  const float* Wu2    = (const float*)d_in[7];
  const float* bn1    = (const float*)d_in[8];
  const float* bs1    = (const float*)d_in[9];
  const float* bu1    = (const float*)d_in[10];
  const float* bn2    = (const float*)d_in[11];
  const float* bs2    = (const float*)d_in[12];
  const float* bu2    = (const float*)d_in[13];
  const float* gamma1 = (const float*)d_in[14];
  const float* beta1  = (const float*)d_in[15];
  const float* gamma2 = (const float*)d_in[16];
  const float* beta2  = (const float*)d_in[17];
  const float* Wp     = (const float*)d_in[18];
  const float* bp     = (const float*)d_in[19];
  const int*   ei0    = (const int*)d_in[20];
  const int*   ei1    = (const int*)d_in[21];
  const int*   ei2    = (const int*)d_in[22];

  const int N = in_sizes[0] / 128;
  const int E = in_sizes[20] / 2;
  float* out = (float*)d_out;

  // workspace carve (~155 MB total)
  uintptr_t cur = (uintptr_t)d_ws;
  auto carve = [&](size_t bytes) -> void* {
    uintptr_t p = (cur + 255) & ~(uintptr_t)255;
    cur = p + bytes;
    return (void*)p;
  };
  float*          agg   = (float*)carve((size_t)N * 128 * 4);
  float*          cnt   = (float*)carve((size_t)N * 4);
  unsigned short* t1b   = (unsigned short*)carve((size_t)N * 128 * 2);
  unsigned short* t2b   = (unsigned short*)carve((size_t)N * 128 * 2);
  float*          hA0   = (float*)carve((size_t)N * 128 * 4);
  float*          hA1   = (float*)carve((size_t)N * 128 * 4);
  float*          hB0   = (float*)carve((size_t)N * 128 * 4);
  float*          hB1   = (float*)carve((size_t)N * 128 * 4);
  float*          stats = (float*)carve(256 * 4);
  unsigned short* wblk  = (unsigned short*)carve((size_t)24 * 16384 * 2);

  unsigned short* Wn1t = wblk +  0 * 16384;  // 3 blocks
  unsigned short* Ws1t = wblk +  3 * 16384;  // 3 blocks
  unsigned short* Wu1t = wblk +  6 * 16384;  // 6 blocks (t0_top,t0_bot,t1_top,...)
  unsigned short* Wn2t = wblk + 12 * 16384;
  unsigned short* Ws2t = wblk + 15 * 16384;
  unsigned short* Wu2t = wblk + 18 * 16384;

  // one-time weight convert+transpose to bf16 [n][k]
  k_convT<<<dim3(3 * 64), dim3(256), 0, stream>>>(Wn1, Wn1t);
  k_convT<<<dim3(3 * 64), dim3(256), 0, stream>>>(Ws1, Ws1t);
  k_convT<<<dim3(6 * 64), dim3(256), 0, stream>>>(Wu1, Wu1t);
  k_convT<<<dim3(3 * 64), dim3(256), 0, stream>>>(Wn2, Wn2t);
  k_convT<<<dim3(3 * 64), dim3(256), 0, stream>>>(Ws2, Ws2t);
  k_convT<<<dim3(6 * 64), dim3(256), 0, stream>>>(Wu2, Wu2t);

  const int gElem = (N * 128 + 255) / 256;
  const int gCnt  = (N + 255) / 256;
  const int gScat = (int)(((long)E * 32 + 255) / 256);
  const int gGemm = (N + 127) / 128;

  auto runLayer = [&](const float* xi0, const float* xi1,
                      const unsigned short* Wnt, const unsigned short* Wst,
                      const unsigned short* Wut,
                      const float* bn, const float* bs, const float* bu,
                      const float* gamma, const float* beta,
                      float* ho0, float* ho1) {
    struct MT { const float* src; const float* dst; const int* ei; float* out; int accum; int t; };
    const MT mts[3] = {
      { xi0, xi1, ei0, ho1, 0, 0 },   // n0 -> n1
      { xi1, xi0, ei1, ho0, 0, 1 },   // n1 -> n0
      { xi0, xi0, ei2, ho0, 1, 2 },   // n0 -> n0 (accumulate)
    };
    for (int i = 0; i < 3; ++i) {
      const MT& m = mts[i];
      k_zero<<<gElem, 256, 0, stream>>>(agg, N * 128);
      k_zero<<<gCnt,  256, 0, stream>>>(cnt, N);
      k_scatter<<<gScat, 256, 0, stream>>>(m.src, m.ei, agg, cnt, E);
      k_finalize<<<gElem, 256, 0, stream>>>(agg, cnt, N * 128);
      k_gemm_xw<<<gGemm, 256, 0, stream>>>(agg,   Wnt + m.t * 16384, bn + m.t * 128, t1b, N);
      k_gemm_xw<<<gGemm, 256, 0, stream>>>(m.dst, Wst + m.t * 16384, bs + m.t * 128, t2b, N);
      k_gemm2<<<gGemm, 256, 0, stream>>>(t1b, t2b,
                                         Wut + (2 * m.t) * 16384, Wut + (2 * m.t + 1) * 16384,
                                         bu + m.t * 128, m.out, N, m.accum);
    }
    // h0 = LeakyReLU(BN(0.5*(m1+m2))) ; h1 = LeakyReLU(BN(m0))
    k_zero<<<1, 256, 0, stream>>>(stats, 256);
    k_bn_stats<<<256, 256, 0, stream>>>(ho0, stats, 0.5f, N);
    k_bn_apply<<<gElem, 256, 0, stream>>>(ho0, stats, gamma + 0, beta + 0, 0.5f, N);
    k_zero<<<1, 256, 0, stream>>>(stats, 256);
    k_bn_stats<<<256, 256, 0, stream>>>(ho1, stats, 1.0f, N);
    k_bn_apply<<<gElem, 256, 0, stream>>>(ho1, stats, gamma + 128, beta + 128, 1.0f, N);
  };

  runLayer(x0,  x1,  Wn1t, Ws1t, Wu1t, bn1, bs1, bu1, gamma1, beta1, hA0, hA1);
  runLayer(hA0, hA1, Wn2t, Ws2t, Wu2t, bn2, bs2, bu2, gamma2, beta2, hB0, hB1);

  const int gProj = (N * 8 + 255) / 256;
  k_proj<<<gProj, 256, 0, stream>>>(hB0, Wp,           bp,     out,               N);
  k_proj<<<gProj, 256, 0, stream>>>(hB1, Wp + 128 * 8, bp + 8, out + (size_t)N * 8, N);
}
